// HadamardClassifier_60962765799486
// MI455X (gfx1250) — compile-verified, matrix operand order checked by reading, not run on hardware
//
#include <hip/hip_runtime.h>

typedef __attribute__((ext_vector_type(16))) _Float16 v16h;
typedef __attribute__((ext_vector_type(8)))  _Float16 v8h;
typedef __attribute__((ext_vector_type(4)))  _Float16 v4h;
typedef __attribute__((ext_vector_type(8)))  float    v8f;
typedef __attribute__((ext_vector_type(4)))  float    v4f;

constexpr int   IN_DIM  = 2048;
constexpr int   OUT_DIM = 16384;
constexpr int   BATCH   = 1024;
constexpr float L2_EPS  = 1e-12f;

constexpr int BN = 128;  // workgroup N tile
constexpr int BM = 64;   // workgroup M tile
constexpr int BK = 32;   // K step == WMMA f16 K
constexpr int LDSB_STRIDE = 40;  // halves per LDS "n" row: 32 K + 8 pad, keeps 16B align

// ---------------- Kernel 1: per-row L2 norm reciprocal --------------------
__global__ __launch_bounds__(256) void hc_rownorm(const float* __restrict__ x,
                                                  float* __restrict__ rnorm) {
  __shared__ float red[8];
  const int row = blockIdx.x;
  const float* xr = x + (size_t)row * IN_DIM;
  float s = 0.f;
  for (int i = threadIdx.x; i < IN_DIM; i += 256) { const float v = xr[i]; s += v * v; }
  for (int off = 16; off > 0; off >>= 1) s += __shfl_down(s, off, 32);
  const int lane = threadIdx.x & 31, w = threadIdx.x >> 5;
  if (lane == 0) red[w] = s;
  __syncthreads();
  if (w == 0) {
    s = (lane < 8) ? red[lane] : 0.f;
    for (int off = 4; off > 0; off >>= 1) s += __shfl_down(s, off, 32);
    if (lane == 0) rnorm[row] = rsqrtf(fmaxf(s, L2_EPS));
  }
}

__device__ __forceinline__ v8f wmma_f16(v16h a, v16h b, v8f c) {
  // D = A(16x32 f16) x B(32x16 f16) + C(16x16 f32)
  return __builtin_amdgcn_wmma_f32_16x16x32_f16(false, a, false, b, (short)0, c,
                                                false, false);
}

// ---------------- Kernel 2: pipelined WMMA GEMM with fused epilogue -------
__global__ __launch_bounds__(256) void hc_gemm(const float* __restrict__ x,
                                               const float* __restrict__ H,
                                               const float* __restrict__ scale,
                                               const float* __restrict__ bias,
                                               const float* __restrict__ rnorm,
                                               float* __restrict__ out) {
  __shared__ __align__(16) _Float16 Bs[BN * LDSB_STRIDE];

  const int tileN = blockIdx.x * BN;
  const int tileM = blockIdx.y * BM;
  const int tid   = threadIdx.x;
  const int lane  = tid & 31;
  const int wave  = tid >> 5;
  const int mw    = wave & 3;   // 4 waves along M
  const int nw    = wave >> 2;  // 2 waves along N
  const int l15   = lane & 15;
  const int kb    = (lane < 16) ? 0 : 8;  // half-wave K base per ISA layout

  // A row owned by this lane (lanes 0-15 and 16-31 carry the same M, different K)
  const int m = tileM + mw * 16 + l15;
  const float rn = rnorm[m];
  const float* xrow = x + (size_t)m * IN_DIM;

  // B staging mapping: 256 threads cover a 32(K) x 128(N) f32 tile
  const int tN = tid & 31;  // n-quad index
  const int tK = tid >> 5;  // base K row
  const float* hbase = H + tileN + tN * 4;

  v8f acc[4] = {};

  // ---- software pipeline: prime tile 0 into regs, converted to f16 ----
  v4f hv[4];
  v4h hh[4];
#pragma unroll
  for (int kk = 0; kk < 4; ++kk)
    hv[kk] = *(const v4f*)(hbase + (size_t)(tK + kk * 8) * OUT_DIM);
#pragma unroll
  for (int kk = 0; kk < 4; ++kk)
#pragma unroll
    for (int j = 0; j < 4; ++j) hh[kk][j] = (_Float16)hv[kk][j];

  for (int k0 = 0; k0 < IN_DIM; k0 += BK) {
    // ---- serialized window: only DS stores of the pre-converted tile ----
#pragma unroll
    for (int kk = 0; kk < 4; ++kk) {
      const int k = tK + kk * 8;
#pragma unroll
      for (int j = 0; j < 4; ++j)
        Bs[(tN * 4 + j) * LDSB_STRIDE + k] = hh[kk][j];
    }
    __syncthreads();  // stores visible before any fragment reads

    // ---- issue next tile's global loads; latency overlaps compute below ----
    const bool more = (k0 + BK) < IN_DIM;
    if (more) {
#pragma unroll
      for (int kk = 0; kk < 4; ++kk)
        hv[kk] = *(const v4f*)(hbase + (size_t)(k0 + BK + tK + kk * 8) * OUT_DIM);
    }
    if (k0 + 2 * BK < IN_DIM)  // stay two slabs ahead in L2 (global_prefetch_b8)
      __builtin_prefetch(hbase + (size_t)(k0 + 2 * BK + tK) * OUT_DIM, 0, 0);

    // ---- A fragment from global, scaled by 1/||x||, split hi+lo f16 ----
    v16h ahi, alo;
    {
      const float* ap = xrow + k0 + kb;
      const v4f x0 = *(const v4f*)(ap);
      const v4f x1 = *(const v4f*)(ap + 4);
      const v4f x2 = *(const v4f*)(ap + 16);
      const v4f x3 = *(const v4f*)(ap + 20);
#pragma unroll
      for (int j = 0; j < 4; ++j) {
        float v;
        v = x0[j] * rn; ahi[j]      = (_Float16)v; alo[j]      = (_Float16)(v - (float)ahi[j]);
        v = x1[j] * rn; ahi[4 + j]  = (_Float16)v; alo[4 + j]  = (_Float16)(v - (float)ahi[4 + j]);
        v = x2[j] * rn; ahi[8 + j]  = (_Float16)v; alo[8 + j]  = (_Float16)(v - (float)ahi[8 + j]);
        v = x3[j] * rn; ahi[12 + j] = (_Float16)v; alo[12 + j] = (_Float16)(v - (float)ahi[12 + j]);
      }
    }

    // ---- B fragments: two aligned b128 LDS reads per 16x32 tile ----
    v16h bf[4];
    const int nl0 = nw * 64 + l15;
#pragma unroll
    for (int t = 0; t < 4; ++t) {
      const _Float16* bp = &Bs[(nl0 + 16 * t) * LDSB_STRIDE + kb];
      const v8h blo = *(const v8h*)(bp);        // K = kb .. kb+7
      const v8h bhi = *(const v8h*)(bp + 16);   // K = kb+16 .. kb+23
#pragma unroll
      for (int j = 0; j < 8; ++j) { bf[t][j] = blo[j]; bf[t][8 + j] = bhi[j]; }
    }

    // ---- 8 WMMAs per K step: hi pass then lo pass (compensated f32) ----
#pragma unroll
    for (int t = 0; t < 4; ++t) acc[t] = wmma_f16(ahi, bf[t], acc[t]);
#pragma unroll
    for (int t = 0; t < 4; ++t) acc[t] = wmma_f16(alo, bf[t], acc[t]);

    // ---- convert next tile after WMMAs are issued (wait lands here) ----
    if (more) {
#pragma unroll
      for (int kk = 0; kk < 4; ++kk)
#pragma unroll
        for (int j = 0; j < 4; ++j) hh[kk][j] = (_Float16)hv[kk][j];
    }
    __syncthreads();  // all fragment reads done before next store phase
  }

  // ---- epilogue: out = -scale * acc + bias ----
  const float negscale = -scale[0];
  const int mo = tileM + mw * 16 + ((lane < 16) ? 0 : 8);  // C/D: VGPR r -> M=r or r+8
#pragma unroll
  for (int t = 0; t < 4; ++t) {
    const int n = tileN + nw * 64 + 16 * t + l15;
    const float bv = bias[n];
#pragma unroll
    for (int rr = 0; rr < 8; ++rr)
      out[(size_t)(mo + rr) * OUT_DIM + n] = fmaf(negscale, acc[t][rr], bv);
  }
}

// ---------------- Harness entry ----------------
extern "C" void kernel_launch(void* const* d_in, const int* in_sizes, int n_in,
                              void* d_out, int out_size, void* d_ws, size_t ws_size,
                              hipStream_t stream) {
  const float* x     = (const float*)d_in[0];  // [1024, 2048]
  const float* scale = (const float*)d_in[1];  // [1]
  const float* bias  = (const float*)d_in[2];  // [16384]
  const float* H     = (const float*)d_in[3];  // [2048, 16384]
  float* out   = (float*)d_out;                // [1024, 16384]
  float* rnorm = (float*)d_ws;                 // BATCH floats of scratch

  hc_rownorm<<<BATCH, 256, 0, stream>>>(x, rnorm);
  dim3 grid(OUT_DIM / BN, BATCH / BM);         // 128 x 16 = 2048 workgroups
  hc_gemm<<<grid, 256, 0, stream>>>(x, H, scale, bias, rnorm, out);

  (void)in_sizes; (void)n_in; (void)out_size; (void)ws_size;
}
